// AttentionGCN_14405320310824
// MI455X (gfx1250) — matrix operand order repaired and checked by reading, not compiled, not run on hardware
//
#include <hip/hip_runtime.h>

#define NB 8
#define NN 10000
#define NF 8
#define NP 12
#define NE 160000
#define NT 64
#define NC 32
#define NPAD 34   // padded row stride (floats) for the LDS H staging tile

typedef __attribute__((ext_vector_type(16))) __bf16 v16bf;
typedef __attribute__((ext_vector_type(8)))  float  v8f;

__device__ __forceinline__ v8f splat8(float x) {
  v8f r;
#pragma unroll
  for (int i = 0; i < 8; ++i) r[i] = x;
  return r;
}

// hardware v_rcp_f32 (1 ulp approx) instead of IEEE division
__device__ __forceinline__ float fast_rcp(float x) {
  return __builtin_amdgcn_rcpf(x);
}

__device__ __forceinline__ float sigmoidf_(float x) {
  return fast_rcp(1.0f + __expf(-x));      // v_exp + v_add + v_rcp
}

__device__ __forceinline__ float tanhf_(float x) {
  x = fminf(fmaxf(x, -15.0f), 15.0f);
  float e = __expf(2.0f * x);
  return (e - 1.0f) * fast_rcp(e + 1.0f);  // v_exp + v_rcp + 2 alu
}

// ---------------- degree / norm ----------------
__global__ void init_deg_kernel(float* deg) {
  int i = blockIdx.x * 256 + threadIdx.x;
  if (i < NN) deg[i] = 1.0f;           // self-loop weight 1
}

__global__ void deg_accum_kernel(const long long* __restrict__ ei,
                                 const float* __restrict__ w,
                                 float* __restrict__ deg) {
  int e = blockIdx.x * 256 + threadIdx.x;
  if (e < NE) {
    int col = (int)ei[NE + e];
    atomicAdd(&deg[col], w[e]);
  }
}

__global__ void deg_rsqrt_kernel(float* deg) {
  int i = blockIdx.x * 256 + threadIdx.x;
  if (i < NN) deg[i] = rsqrtf(deg[i]); // deg >= 1 always
}

// ---------------- aggregation: Xagg[b,n,p,f] = sum_edges + self loop -------
// x layout: (B,N,F,P); Xagg layout (B,N,P,F) so the 8 per-period features are
// contiguous for the GRU kernel.
__global__ void init_xagg_kernel(const float* __restrict__ x,
                                 const float* __restrict__ dis,
                                 float* __restrict__ xagg) {
  size_t i = (size_t)blockIdx.x * 256 + threadIdx.x;
  if (i < (size_t)NB * NN * NF * NP) {
    int k = (int)(i % (NF * NP));      // k = f*NP + p  (read-coalesced)
    size_t bn = i / (NF * NP);
    int n = (int)(bn % NN);
    int f = k / NP, p = k % NP;
    float d = dis[n];
    xagg[bn * (NF * NP) + p * NF + f] = d * d * x[i];
  }
}

__global__ __launch_bounds__(256) void aggregate_kernel(
    const float* __restrict__ x, const long long* __restrict__ ei,
    const float* __restrict__ w, const float* __restrict__ dis,
    float* __restrict__ xagg) {
  int e = blockIdx.x;                          // one edge per block
  int row = (int)ei[e];
  int col = (int)ei[NE + e];
  float coeff = dis[row] * w[e] * dis[col];
  for (int j = threadIdx.x; j < NB * NF * NP; j += 256) {
    int b = j / (NF * NP);
    int k = j % (NF * NP);                     // k = f*NP + p
    int f = k / NP, p = k % NP;
    float v = x[(size_t)b * NN * NF * NP + (size_t)row * NF * NP + k];
    atomicAdd(&xagg[(size_t)b * NN * NF * NP + (size_t)col * NF * NP + p * NF + f],
              coeff * v);
  }
}

// ---------------- fold weights: A_g = W_g @ L_g[:32]; bias; softmax(att) ----
__global__ __launch_bounds__(256) void prep_kernel(
    const float* __restrict__ Wz, const float* __restrict__ Wr,
    const float* __restrict__ Wh, const float* __restrict__ bz,
    const float* __restrict__ br, const float* __restrict__ bh,
    const float* __restrict__ Lz, const float* __restrict__ Lr,
    const float* __restrict__ Lh, const float* __restrict__ Lzb,
    const float* __restrict__ Lrb, const float* __restrict__ Lhb,
    const float* __restrict__ att, float* __restrict__ Am,
    float* __restrict__ cb, float* __restrict__ probs) {
  const float* W[3] = {Wz, Wr, Wh};
  const float* L[3] = {Lz, Lr, Lh};
  const float* bb[3] = {bz, br, bh};
  const float* Lb[3] = {Lzb, Lrb, Lhb};
  int tid = threadIdx.x;
  int f = tid / NC, c = tid % NC;              // f<8, c<32; 256 threads exactly
  for (int g = 0; g < 3; ++g) {
    float a = 0.0f;
    for (int k = 0; k < NC; ++k) a += W[g][f * NC + k] * L[g][k * NC + c];
    Am[g * 256 + tid] = a;
  }
  if (tid < NC) {
    for (int g = 0; g < 3; ++g) {
      float a = Lb[g][tid];
      for (int k = 0; k < NC; ++k) a += bb[g][k] * L[g][k * NC + tid];
      cb[g * NC + tid] = a;
    }
  }
  if (tid == 0) {
    float m = -1e30f;
    for (int p = 0; p < NP; ++p) m = fmaxf(m, att[p]);
    float s = 0.0f, ex[NP];
    for (int p = 0; p < NP; ++p) { ex[p] = __expf(att[p] - m); s += ex[p]; }
    for (int p = 0; p < NP; ++p) probs[p] = ex[p] * fast_rcp(s);
  }
}

// ---------------- WMMA GRU recurrence -------------------------------------
// A-fragment (16x32 bf16) from LDS H staging tile (row-major, NPAD stride).
__device__ __forceinline__ v16bf load_afrag(const float* Hs, int lane) {
  int row = lane & 15;
  int base = (lane < 16) ? 0 : 8;
  const float* hp = Hs + row * NPAD;
  v16bf a;
#pragma unroll
  for (int i = 0; i < 8; ++i)  a[i] = (__bf16)hp[base + i];        // K = base+i
#pragma unroll
  for (int i = 8; i < 16; ++i) a[i] = (__bf16)hp[base + 8 + i];    // K = base+16..
  return a;
}

// Store a 16x32 f32 tile (C/D layout, two 16x16 halves) into the staging tile.
__device__ __forceinline__ void store_tile(float* Hs, int lane, v8f t0, v8f t1) {
  int n = lane & 15;
  int mb = (lane < 16) ? 0 : 8;
#pragma unroll
  for (int v = 0; v < 8; ++v) {
    Hs[(mb + v) * NPAD + n]      = t0[v];
    Hs[(mb + v) * NPAD + 16 + n] = t1[v];
  }
}

__global__ __launch_bounds__(128) void gru_kernel(
    const float* __restrict__ xagg, const float* __restrict__ Am,
    const float* __restrict__ cb, const float* __restrict__ probs,
    const float* __restrict__ Lz, const float* __restrict__ Lr,
    const float* __restrict__ Lh, const float* __restrict__ lin1_w,
    const float* __restrict__ lin1_b, float* __restrict__ y) {
  __shared__ v16bf Bfrag[3][2][2][32];    // [gate][ntile][kblock][lane]
  __shared__ float Hst[4][16 * NPAD];     // per-wave staging tile
  __shared__ float s_l1[NC * NP];
  __shared__ float s_cb[3 * NC];
  __shared__ float s_pr[NP];
  __shared__ float s_l1b[NP];

  const float* Lw[3] = {Lz, Lr, Lh};
  int tid = threadIdx.x;

  // Build B-operand fragments:
  //  kb==0: rows K=0..31 are U_g = L_g[32:64]; kb==1: rows 0..7 are A_g, rest 0.
  //  B layout: lane l -> column c=j*16+(l&15); elem i -> K=8*(l>>4)+i (i<8) / 16+8*(l>>4)+(i-8).
  for (int idx = tid; idx < 3 * 2 * 2 * 32 * 16; idx += 128) {
    int e  = idx & 15;
    int l  = (idx >> 4) & 31;
    int kb = (idx >> 9) & 1;
    int j  = (idx >> 10) & 1;
    int g  = idx >> 11;
    int c  = j * 16 + (l & 15);
    float v;
    if (kb == 0) {
      int K = (e < 8) ? (8 * (l >> 4) + e) : (16 + 8 * (l >> 4) + (e - 8));
      v = Lw[g][(NC + K) * NC + c];
    } else {
      v = (l < 16 && e < 8) ? Am[g * 256 + e * NC + c] : 0.0f;
    }
    Bfrag[g][j][kb][l][e] = (__bf16)v;
  }
  for (int i = tid; i < NC * NP; i += 128) s_l1[i] = lin1_w[i];
  if (tid < 3 * NC) s_cb[tid] = cb[tid];
  if (tid < NP) { s_pr[tid] = probs[tid]; s_l1b[tid] = lin1_b[tid]; }
  __syncthreads();

  int wave = tid >> 5, lane = tid & 31;
  float* Hs = Hst[wave];
  int tile = blockIdx.x * 4 + wave;       // 5000 tiles of 16 (b,n) pairs
  int ubase = tile * 16;
  int cl = lane & 15;

  float bz0 = s_cb[cl],          bz1 = s_cb[16 + cl];
  float br0 = s_cb[NC + cl],     br1 = s_cb[NC + 16 + cl];
  float bh0 = s_cb[2 * NC + cl], bh1 = s_cb[2 * NC + 16 + cl];

  v8f H0 = splat8(0.0f), H1 = splat8(0.0f);       // H in C/D layout
  v8f Ha0 = splat8(0.0f), Ha1 = splat8(0.0f);     // H_accum
  store_tile(Hs, lane, H0, H1);

  const float* xg = xagg + (size_t)ubase * (NF * NP);

#pragma unroll 1
  for (int p = 0; p < NP; ++p) {
    // x A-fragment: lanes 0..15 hold row's 8 features at K'=0..7, rest zero
    v16bf ax;
#pragma unroll
    for (int i = 0; i < 16; ++i) ax[i] = (__bf16)0.0f;
    if (lane < 16) {
      const float* xp = xg + lane * (NF * NP) + p * NF;
#pragma unroll
      for (int i = 0; i < NF; ++i) ax[i] = (__bf16)xp[i];
    }
    v16bf aH = load_afrag(Hs, lane);

    v8f z0 = splat8(bz0), z1 = splat8(bz1);
    z0 = __builtin_amdgcn_wmma_f32_16x16x32_bf16(false, aH, false, Bfrag[0][0][0][lane], (short)0, z0, false, false);
    z0 = __builtin_amdgcn_wmma_f32_16x16x32_bf16(false, ax, false, Bfrag[0][0][1][lane], (short)0, z0, false, false);
    z1 = __builtin_amdgcn_wmma_f32_16x16x32_bf16(false, aH, false, Bfrag[0][1][0][lane], (short)0, z1, false, false);
    z1 = __builtin_amdgcn_wmma_f32_16x16x32_bf16(false, ax, false, Bfrag[0][1][1][lane], (short)0, z1, false, false);

    v8f r0 = splat8(br0), r1 = splat8(br1);
    r0 = __builtin_amdgcn_wmma_f32_16x16x32_bf16(false, aH, false, Bfrag[1][0][0][lane], (short)0, r0, false, false);
    r0 = __builtin_amdgcn_wmma_f32_16x16x32_bf16(false, ax, false, Bfrag[1][0][1][lane], (short)0, r0, false, false);
    r1 = __builtin_amdgcn_wmma_f32_16x16x32_bf16(false, aH, false, Bfrag[1][1][0][lane], (short)0, r1, false, false);
    r1 = __builtin_amdgcn_wmma_f32_16x16x32_bf16(false, ax, false, Bfrag[1][1][1][lane], (short)0, r1, false, false);

#pragma unroll
    for (int i = 0; i < 8; ++i) {
      z0[i] = sigmoidf_(z0[i]); z1[i] = sigmoidf_(z1[i]);
      r0[i] = sigmoidf_(r0[i]); r1[i] = sigmoidf_(r1[i]);
    }

    // H * R -> staging -> A-fragment for the candidate gate
    v8f hr0, hr1;
#pragma unroll
    for (int i = 0; i < 8; ++i) { hr0[i] = H0[i] * r0[i]; hr1[i] = H1[i] * r1[i]; }
    store_tile(Hs, lane, hr0, hr1);
    v16bf aR = load_afrag(Hs, lane);

    v8f t0 = splat8(bh0), t1 = splat8(bh1);
    t0 = __builtin_amdgcn_wmma_f32_16x16x32_bf16(false, aR, false, Bfrag[2][0][0][lane], (short)0, t0, false, false);
    t0 = __builtin_amdgcn_wmma_f32_16x16x32_bf16(false, ax, false, Bfrag[2][0][1][lane], (short)0, t0, false, false);
    t1 = __builtin_amdgcn_wmma_f32_16x16x32_bf16(false, aR, false, Bfrag[2][1][0][lane], (short)0, t1, false, false);
    t1 = __builtin_amdgcn_wmma_f32_16x16x32_bf16(false, ax, false, Bfrag[2][1][1][lane], (short)0, t1, false, false);

    float pw = s_pr[p];
#pragma unroll
    for (int i = 0; i < 8; ++i) {
      t0[i] = tanhf_(t0[i]); t1[i] = tanhf_(t1[i]);
      // H = Ht + z*(H - Ht)  (== z*H + (1-z)*Ht)
      H0[i] = t0[i] + z0[i] * (H0[i] - t0[i]);
      H1[i] = t1[i] + z1[i] * (H1[i] - t1[i]);
      Ha0[i] += pw * H0[i];  Ha1[i] += pw * H1[i];
    }
    store_tile(Hs, lane, H0, H1);   // for next period's A-fragment
  }

  // Epilogue: relu(H_accum) @ lin1_w + lin1_b  -> y[b][p][n]
#pragma unroll
  for (int i = 0; i < 8; ++i) {
    Ha0[i] = fmaxf(Ha0[i], 0.0f);
    Ha1[i] = fmaxf(Ha1[i], 0.0f);
  }
  store_tile(Hs, lane, Ha0, Ha1);
  if (lane < 16) {
    int u = ubase + lane;
    int b = u / NN, n = u % NN;
    const float* hrow = Hs + lane * NPAD;
#pragma unroll
    for (int q = 0; q < NP; ++q) {
      float acc = s_l1b[q];
#pragma unroll
      for (int c = 0; c < NC; ++c) acc += hrow[c] * s_l1[c * NP + q];
      y[(size_t)b * NP * NN + (size_t)q * NN + n] = acc;
    }
  }
}

// ---------------- final reduction: out[b,t,p] = sum_n y[b,p,n]*W2[n,t]+b2 --
__global__ __launch_bounds__(256) void out_kernel(
    const float* __restrict__ y, const float* __restrict__ W2,
    const float* __restrict__ b2, float* __restrict__ out) {
  __shared__ float red[256];
  int b = blockIdx.x / NP, p = blockIdx.x % NP;
  int t = threadIdx.x & 63, s = threadIdx.x >> 6;
  const float* yp = y + (size_t)b * NP * NN + (size_t)p * NN;
  float acc = 0.0f;
  for (int n = s; n < NN; n += 4) acc += yp[n] * W2[(size_t)n * NT + t];
  red[threadIdx.x] = acc;
  __syncthreads();
  if (s == 0) {
    float total = red[t] + red[64 + t] + red[128 + t] + red[192 + t] + b2[t];
    out[(size_t)b * NT * NP + (size_t)t * NP + p] = total;
  }
}

extern "C" void kernel_launch(void* const* d_in, const int* in_sizes, int n_in,
                              void* d_out, int out_size, void* d_ws, size_t ws_size,
                              hipStream_t stream) {
  const float*     x    = (const float*)d_in[0];
  const long long* ei   = (const long long*)d_in[1];
  const float*     ew   = (const float*)d_in[2];
  const float*     Wz   = (const float*)d_in[3];
  const float*     bz   = (const float*)d_in[4];
  const float*     Wr   = (const float*)d_in[5];
  const float*     br   = (const float*)d_in[6];
  const float*     Wh   = (const float*)d_in[7];
  const float*     bh   = (const float*)d_in[8];
  const float*     Lz   = (const float*)d_in[9];
  const float*     Lzb  = (const float*)d_in[10];
  const float*     Lr   = (const float*)d_in[11];
  const float*     Lrb  = (const float*)d_in[12];
  const float*     Lh   = (const float*)d_in[13];
  const float*     Lhb  = (const float*)d_in[14];
  const float*     att  = (const float*)d_in[15];
  const float*     l1w  = (const float*)d_in[16];
  const float*     l1b  = (const float*)d_in[17];
  const float*     l2w  = (const float*)d_in[18];
  const float*     l2b  = (const float*)d_in[19];
  float* out = (float*)d_out;

  // workspace layout (floats)
  float* ws     = (float*)d_ws;
  float* d_deg  = ws;                        // 10240
  float* d_xagg = d_deg + 10240;             // 7,680,000
  float* d_Am   = d_xagg + (size_t)NB * NN * NF * NP;   // 768
  float* d_cb   = d_Am + 768;                // 96
  float* d_pr   = d_cb + 96;                 // 16
  float* d_y    = d_pr + 16;                 // 960,000

  init_deg_kernel<<<(NN + 255) / 256, 256, 0, stream>>>(d_deg);
  deg_accum_kernel<<<(NE + 255) / 256, 256, 0, stream>>>(ei, ew, d_deg);
  deg_rsqrt_kernel<<<(NN + 255) / 256, 256, 0, stream>>>(d_deg);

  size_t total = (size_t)NB * NN * NF * NP;
  init_xagg_kernel<<<(unsigned)((total + 255) / 256), 256, 0, stream>>>(x, d_deg, d_xagg);
  aggregate_kernel<<<NE, 256, 0, stream>>>(x, ei, ew, d_deg, d_xagg);

  prep_kernel<<<1, 256, 0, stream>>>(Wz, Wr, Wh, bz, br, bh, Lz, Lr, Lh,
                                     Lzb, Lrb, Lhb, att, d_Am, d_cb, d_pr);

  gru_kernel<<<(NB * NN) / (16 * 4), 128, 0, stream>>>(
      d_xagg, d_Am, d_cb, d_pr, Lz, Lr, Lh, l1w, l1b, d_y);

  out_kernel<<<NB * NP, 256, 0, stream>>>(d_y, l2w, l2b, out);
}
